// CrossFormerBlock_66116726555298
// MI455X (gfx1250) — matrix operand order repaired and checked by reading, not compiled, not run on hardware
//
#include <hip/hip_runtime.h>
#include <hip/hip_bf16.h>
#include <math.h>

typedef __bf16 bf16;
typedef __attribute__((ext_vector_type(16))) __bf16 v16bf;
typedef __attribute__((ext_vector_type(8)))  __bf16 v8bf;
typedef __attribute__((ext_vector_type(8)))  float  v8f;

#define ROWS   100352      // 32 * 56 * 56
#define CCH    256
#define CHID   1024

// ---------------------------------------------------------------------------
// CDNA5 async copy: global -> LDS, 16B per lane, tracked by ASYNCcnt.
// ---------------------------------------------------------------------------
__device__ __forceinline__ void async_copy_b128(unsigned lds_off, const void* gaddr) {
    asm volatile("global_load_async_to_lds_b128 %0, %1, off"
                 :: "v"(lds_off), "v"(gaddr) : "memory");
}
template <int N>
__device__ __forceinline__ void wait_async() {
    asm volatile("s_wait_asynccnt %0" :: "i"(N) : "memory");
}

// ---------------------------------------------------------------------------
// fp32 -> bf16 with transpose: in [K,N] row-major -> out [N,K] row-major (B^T)
// ---------------------------------------------------------------------------
__global__ void f32_to_bf16_tr_kernel(const float* __restrict__ in,
                                      bf16* __restrict__ out, int K, int N) {
    int i = blockIdx.x * 256 + threadIdx.x;
    if (i < K * N) {
        int k = i / N, n = i - k * N;
        out[(size_t)n * K + k] = (bf16)in[i];
    }
}

// ---------------------------------------------------------------------------
// LayerNorm over C=256 -> bf16. One wave32 per row, 8 rows per block.
// ---------------------------------------------------------------------------
__global__ __launch_bounds__(256)
void ln_bf16_kernel(const float* __restrict__ in, const float* __restrict__ g,
                    const float* __restrict__ b, bf16* __restrict__ out, int rows) {
    const int wave = threadIdx.x >> 5;
    const int lane = threadIdx.x & 31;
    const int row  = blockIdx.x * 8 + wave;
    if (row >= rows) return;
    const float* p = in + (size_t)row * CCH;
    float v[8];
    float s = 0.f, s2 = 0.f;
    #pragma unroll
    for (int i = 0; i < 8; ++i) {
        v[i] = p[lane + i * 32];
        s += v[i]; s2 += v[i] * v[i];
    }
    #pragma unroll
    for (int off = 16; off > 0; off >>= 1) {
        s  += __shfl_xor(s,  off, 32);
        s2 += __shfl_xor(s2, off, 32);
    }
    const float mean = s * (1.f / 256.f);
    const float var  = s2 * (1.f / 256.f) - mean * mean;
    const float inv  = rsqrtf(var + 1e-5f);
    bf16* o = out + (size_t)row * CCH;
    #pragma unroll
    for (int i = 0; i < 8; ++i) {
        int c = lane + i * 32;
        o[c] = (bf16)((v[i] - mean) * inv * g[c] + b[c]);
    }
}

// ---------------------------------------------------------------------------
// DynamicPosBias MLP: 169 bias offsets -> posT[169][8].
// ---------------------------------------------------------------------------
__device__ __forceinline__ void ln16_relu(const float* in, const float* g,
                                          const float* b, float* out) {
    float s = 0.f;
    #pragma unroll
    for (int i = 0; i < 16; ++i) s += in[i];
    const float mu = s * (1.f / 16.f);
    float v = 0.f;
    #pragma unroll
    for (int i = 0; i < 16; ++i) { float d = in[i] - mu; v += d * d; }
    const float inv = rsqrtf(v * (1.f / 16.f) + 1e-5f);
    #pragma unroll
    for (int i = 0; i < 16; ++i) {
        float t = (in[i] - mu) * inv * g[i] + b[i];
        out[i] = fmaxf(t, 0.f);
    }
}

__global__ void posbias_kernel(const float* __restrict__ pw, const float* __restrict__ pb,
                               const float* g1, const float* b1, const float* w1, const float* wb1,
                               const float* g2, const float* b2, const float* w2, const float* wb2,
                               const float* g3, const float* b3, const float* w3, const float* wb3,
                               float* __restrict__ posT) {
    const int t = threadIdx.x;
    if (t >= 169) return;
    const float bh = (float)(t / 13) - 6.f;
    const float bw = (float)(t % 13) - 6.f;
    float a[16], r[16], c[16];
    #pragma unroll
    for (int j = 0; j < 16; ++j) a[j] = bh * pw[j] + bw * pw[16 + j] + pb[j];
    ln16_relu(a, g1, b1, r);
    for (int j = 0; j < 16; ++j) { float s = wb1[j]; for (int i = 0; i < 16; ++i) s += r[i] * w1[i * 16 + j]; c[j] = s; }
    ln16_relu(c, g2, b2, r);
    for (int j = 0; j < 16; ++j) { float s = wb2[j]; for (int i = 0; i < 16; ++i) s += r[i] * w2[i * 16 + j]; a[j] = s; }
    ln16_relu(a, g3, b3, r);
    for (int h = 0; h < 8; ++h) { float s = wb3[h]; for (int i = 0; i < 16; ++i) s += r[i] * w3[i * 8 + h]; posT[t * 8 + h] = s; }
}

// ---------------------------------------------------------------------------
// Windowed attention: one 64-thread block per (window, head).
// Output rows written in UNGROUPED (original token) order.
// ---------------------------------------------------------------------------
__global__ __launch_bounds__(64)
void attn_kernel(const bf16* __restrict__ qkv, const float* __restrict__ posT,
                 bf16* __restrict__ out) {
    __shared__ float Ks[49][32];
    __shared__ float Vs[49][32];
    const int head = blockIdx.x;
    const int wf   = blockIdx.y;            // flat window id: b*64 + wi
    const int b    = wf >> 6;
    const int wi   = wf & 63;
    const int wh   = wi >> 3, ww = wi & 7;
    const int t    = threadIdx.x;

    if (t < 49) {
        const int hh = wh * 7 + t / 7, wp = ww * 7 + t % 7;
        const size_t row = (size_t)b * 3136 + hh * 56 + wp;
        const bf16* kp = qkv + row * 768 + 256 + head * 32;
        const bf16* vp = qkv + row * 768 + 512 + head * 32;
        #pragma unroll
        for (int d = 0; d < 32; ++d) { Ks[t][d] = (float)kp[d]; Vs[t][d] = (float)vp[d]; }
    }
    __syncthreads();
    if (t >= 49) return;

    const int ph = t / 7, pw = t % 7;
    const int hh = wh * 7 + ph, wp = ww * 7 + pw;
    const size_t row = (size_t)b * 3136 + hh * 56 + wp;
    const bf16* qp = qkv + row * 768 + head * 32;
    float q[32];
    #pragma unroll
    for (int d = 0; d < 32; ++d) q[d] = (float)qp[d] * 0.17677669529f; // 1/sqrt(32)

    float s[49];
    float mx = -1e30f;
    for (int j = 0; j < 49; ++j) {
        float dot = 0.f;
        #pragma unroll
        for (int d = 0; d < 32; ++d) dot += q[d] * Ks[j][d];
        const int dh = ph - j / 7 + 6;
        const int dw = pw - j % 7 + 6;
        dot += posT[(dh * 13 + dw) * 8 + head];
        s[j] = dot;
        mx = fmaxf(mx, dot);
    }
    float denom = 0.f;
    float o[32];
    #pragma unroll
    for (int d = 0; d < 32; ++d) o[d] = 0.f;
    for (int j = 0; j < 49; ++j) {
        const float p = __expf(s[j] - mx);
        denom += p;
        #pragma unroll
        for (int d = 0; d < 32; ++d) o[d] += p * Vs[j][d];
    }
    const float rinv = 1.f / denom;
    bf16* op = out + row * 256 + head * 32;
    #pragma unroll
    for (int d = 0; d < 32; ++d) op[d] = (bf16)(o[d] * rinv);
}

// ---------------------------------------------------------------------------
// bf16 WMMA GEMM: C[M,N] = A[M,K] @ B[K,N], with B supplied TRANSPOSED
// (Bt = B^T, [N,K] row-major). Block tile 128x128, BK=32, 8 waves in a
// 4(M) x 2(N) grid, each wave = 2x4 v_wmma_f32_16x16x32_bf16 accumulators.
// Double-buffered LDS fed by GLOBAL_LOAD_ASYNC_TO_LDS_B128 (ASYNCcnt).
// ---------------------------------------------------------------------------
enum { EPI_BF16 = 0, EPI_GELU_BF16 = 1, EPI_RES_F32 = 2 };

template <int EPI>
__global__ __launch_bounds__(256)
void gemm_bf16_kernel(const bf16* __restrict__ A, const bf16* __restrict__ Bt,
                      const float* __restrict__ bias, const float* __restrict__ res,
                      void* __restrict__ out, int M, int N, int K) {
    __shared__ bf16 As[2][128][40];   // [m][k], +8 pad: 16B-aligned rows, no conflicts
    __shared__ bf16 Bs[2][128][40];   // [n][k]

    const int tid   = threadIdx.x;
    const int lane  = tid & 31;
    const int wave  = tid >> 5;
    const int wm    = wave & 3;      // 4 waves along M (32 rows each)
    const int wn    = wave >> 2;     // 2 waves along N (64 cols each)
    const int m0    = blockIdx.y * 128;
    const int n0    = blockIdx.x * 128;
    const int l16   = lane & 15;
    const int khalf = lane >> 4;

    const int row2 = tid >> 1;          // 0..127: tile row this thread stages
    const int col2 = (tid & 1) * 16;    // 0 or 16: K-offset within the 32-wide slab

    const int niter = K >> 5;

    v8f acc[2][4];
    #pragma unroll
    for (int mt = 0; mt < 2; ++mt)
        #pragma unroll
        for (int nt = 0; nt < 4; ++nt)
            #pragma unroll
            for (int r = 0; r < 8; ++r) acc[mt][nt][r] = 0.f;

    auto prefetch = [&](int stage) {
        const int buf = stage & 1;
        const bf16* ap = A  + (size_t)(m0 + row2) * K + stage * 32 + col2;
        const bf16* bp = Bt + (size_t)(n0 + row2) * K + stage * 32 + col2;
        async_copy_b128((unsigned)(size_t)(void*)&As[buf][row2][col2],     ap);
        async_copy_b128((unsigned)(size_t)(void*)&As[buf][row2][col2 + 8], ap + 8);
        async_copy_b128((unsigned)(size_t)(void*)&Bs[buf][row2][col2],     bp);
        async_copy_b128((unsigned)(size_t)(void*)&Bs[buf][row2][col2 + 8], bp + 8);
    };

    prefetch(0);
    if (niter > 1) prefetch(1);

    for (int i = 0; i < niter; ++i) {
        // Retire the oldest in-flight stage (4 asyncs/wave/stage, in-order).
        if (i + 1 < niter) wait_async<4>();
        else               wait_async<0>();
        __syncthreads();

        const int buf = i & 1;
        v16bf af[2], bfr[4];
        #pragma unroll
        for (int mt = 0; mt < 2; ++mt) {
            const int r = wm * 32 + mt * 16 + l16;
            v8bf lo = *(const v8bf*)&As[buf][r][khalf * 8];        // K 0..7 | 8..15
            v8bf hi = *(const v8bf*)&As[buf][r][16 + khalf * 8];   // K 16..23 | 24..31
            af[mt] = __builtin_shufflevector(lo, hi, 0, 1, 2, 3, 4, 5, 6, 7,
                                             8, 9, 10, 11, 12, 13, 14, 15);
        }
        #pragma unroll
        for (int nt = 0; nt < 4; ++nt) {
            const int c = wn * 64 + nt * 16 + l16;
            v8bf lo = *(const v8bf*)&Bs[buf][c][khalf * 8];        // K 0..7 | 8..15
            v8bf hi = *(const v8bf*)&Bs[buf][c][16 + khalf * 8];   // K 16..23 | 24..31
            bfr[nt] = __builtin_shufflevector(lo, hi, 0, 1, 2, 3, 4, 5, 6, 7,
                                              8, 9, 10, 11, 12, 13, 14, 15);
        }
        #pragma unroll
        for (int mt = 0; mt < 2; ++mt)
            #pragma unroll
            for (int nt = 0; nt < 4; ++nt)
                acc[mt][nt] = __builtin_amdgcn_wmma_f32_16x16x32_bf16(
                    false, af[mt], false, bfr[nt], (short)0, acc[mt][nt],
                    false, false);
        __syncthreads();   // all reads of buf done -> safe to refill it

        if (i + 2 < niter) prefetch(i + 2);
    }

    // --- epilogue: C layout VGPR r -> M = r + 8*khalf, N = lane%16 ---
    #pragma unroll
    for (int mt = 0; mt < 2; ++mt)
        #pragma unroll
        for (int nt = 0; nt < 4; ++nt)
            #pragma unroll
            for (int r = 0; r < 8; ++r) {
                const int grow = m0 + wm * 32 + mt * 16 + khalf * 8 + r;
                const int gcol = n0 + wn * 64 + nt * 16 + l16;
                const size_t idx = (size_t)grow * N + gcol;
                float v = acc[mt][nt][r] + bias[gcol];
                if (EPI == EPI_GELU_BF16) {
                    v = 0.5f * v * (1.f + erff(v * 0.70710678118f));
                    ((bf16*)out)[idx] = (bf16)v;
                } else if (EPI == EPI_RES_F32) {
                    ((float*)out)[idx] = v + res[idx];
                } else {
                    ((bf16*)out)[idx] = (bf16)v;
                }
            }
}

// ---------------------------------------------------------------------------
extern "C" void kernel_launch(void* const* d_in, const int* in_sizes, int n_in,
                              void* d_out, int out_size, void* d_ws, size_t ws_size,
                              hipStream_t stream) {
    const float* x      = (const float*)d_in[0];
    const float* n1g    = (const float*)d_in[1];
    const float* n1b    = (const float*)d_in[2];
    const float* qkv_w  = (const float*)d_in[3];
    const float* qkv_b  = (const float*)d_in[4];
    const float* proj_w = (const float*)d_in[5];
    const float* proj_b = (const float*)d_in[6];
    const float* pos_pw = (const float*)d_in[7];
    const float* pos_pb = (const float*)d_in[8];
    const float* p1g = (const float*)d_in[9];  const float* p1b  = (const float*)d_in[10];
    const float* p1w = (const float*)d_in[11]; const float* p1bb = (const float*)d_in[12];
    const float* p2g = (const float*)d_in[13]; const float* p2b  = (const float*)d_in[14];
    const float* p2w = (const float*)d_in[15]; const float* p2bb = (const float*)d_in[16];
    const float* p3g = (const float*)d_in[17]; const float* p3b  = (const float*)d_in[18];
    const float* p3w = (const float*)d_in[19]; const float* p3bb = (const float*)d_in[20];
    const float* n2g = (const float*)d_in[21]; const float* n2b  = (const float*)d_in[22];
    const float* fc1_w = (const float*)d_in[23]; const float* fc1_b = (const float*)d_in[24];
    const float* fc2_w = (const float*)d_in[25]; const float* fc2_b = (const float*)d_in[26];

    // workspace layout
    char* ws = (char*)d_ws;
    size_t off = 0;
    auto take = [&](size_t bytes) -> char* {
        char* p = ws + off;
        off += (bytes + 255) & ~(size_t)255;
        return p;
    };
    float* posT  = (float*)take(169 * 8 * sizeof(float));
    bf16* wqT    = (bf16*)take((size_t)256 * 768 * 2);     // qkv_w^T  [768,256]
    bf16* wpT    = (bf16*)take((size_t)256 * 256 * 2);     // proj_w^T [256,256]
    bf16* w1T    = (bf16*)take((size_t)256 * 1024 * 2);    // fc1_w^T  [1024,256]
    bf16* w2T    = (bf16*)take((size_t)1024 * 256 * 2);    // fc2_w^T  [256,1024]
    bf16* bufA   = (bf16*)take((size_t)ROWS * 256 * 2);    // LN1 out / attn out / LN2 out
    bf16* bufBig = (bf16*)take((size_t)ROWS * 1024 * 2);   // QKV (154MB), later fc1 act (205MB)
    float* x1    = (float*)take((size_t)ROWS * 256 * 4);   // shortcut + attn branch

    // 1. weight conversions fp32 -> bf16, transposed (GEMM takes B^T)
    { int n = 256 * 768;  f32_to_bf16_tr_kernel<<<(n + 255) / 256, 256, 0, stream>>>(qkv_w,  wqT, 256, 768);  }
    { int n = 256 * 256;  f32_to_bf16_tr_kernel<<<(n + 255) / 256, 256, 0, stream>>>(proj_w, wpT, 256, 256);  }
    { int n = 256 * 1024; f32_to_bf16_tr_kernel<<<(n + 255) / 256, 256, 0, stream>>>(fc1_w,  w1T, 256, 1024); }
    { int n = 1024 * 256; f32_to_bf16_tr_kernel<<<(n + 255) / 256, 256, 0, stream>>>(fc2_w,  w2T, 1024, 256); }

    // 2. dynamic position-bias MLP -> posT[169][8]
    posbias_kernel<<<1, 192, 0, stream>>>(pos_pw, pos_pb, p1g, p1b, p1w, p1bb,
                                          p2g, p2b, p2w, p2bb, p3g, p3b, p3w, p3bb, posT);

    // 3. LN1 -> bf16
    ln_bf16_kernel<<<ROWS / 8, 256, 0, stream>>>(x, n1g, n1b, bufA, ROWS);

    // 4. QKV GEMM: [ROWS,256] @ [256,768] + bias -> bf16
    gemm_bf16_kernel<EPI_BF16><<<dim3(768 / 128, ROWS / 128), 256, 0, stream>>>(
        bufA, wqT, qkv_b, nullptr, bufBig, ROWS, 768, 256);

    // 5. windowed attention (writes ungrouped rows) -> bf16 [ROWS,256]
    attn_kernel<<<dim3(8, 2048), 64, 0, stream>>>(bufBig, posT, bufA);

    // 6. proj GEMM + shortcut residual -> x1 (f32)
    gemm_bf16_kernel<EPI_RES_F32><<<dim3(256 / 128, ROWS / 128), 256, 0, stream>>>(
        bufA, wpT, proj_b, x, x1, ROWS, 256, 256);

    // 7. LN2 -> bf16
    ln_bf16_kernel<<<ROWS / 8, 256, 0, stream>>>(x1, n2g, n2b, bufA, ROWS);

    // 8. fc1 GEMM + exact GELU -> bf16 [ROWS,1024]
    gemm_bf16_kernel<EPI_GELU_BF16><<<dim3(1024 / 128, ROWS / 128), 256, 0, stream>>>(
        bufA, w1T, fc1_b, nullptr, bufBig, ROWS, 1024, 256);

    // 9. fc2 GEMM + residual x1 -> d_out (f32)
    gemm_bf16_kernel<EPI_RES_F32><<<dim3(256 / 128, ROWS / 128), 256, 0, stream>>>(
        bufBig, w2T, fc2_b, x1, (float*)d_out, ROWS, 256, 1024);
}
